// ParameterizationNet_34205119545431
// MI455X (gfx1250) — compile-verified, matrix operand order checked by reading, not run on hardware
//
#include <hip/hip_runtime.h>
#include <hip/hip_bf16.h>

typedef _Float16 v8h  __attribute__((ext_vector_type(8)));
typedef _Float16 v16h __attribute__((ext_vector_type(16)));
typedef float    v8f  __attribute__((ext_vector_type(8)));
typedef float    v4f  __attribute__((ext_vector_type(4)));

#define NI   8192
#define NBD  2048
#define KNN  32
#define R2C  0.01f
#define EPSV 1e-5f

// ---------------- top-k insertion: register-cached threshold, scratch arrays ----
__device__ __forceinline__ void topk_insert(float d, int i, float* bd, int* bi,
                                            int& cnt, float& worst) {
  if (cnt == KNN) {
    if (d >= worst) return;               // common case: pure register compare
    int p = KNN - 1;
    while (p > 0 && bd[p-1] > d) { bd[p] = bd[p-1]; bi[p] = bi[p-1]; --p; }
    bd[p] = d; bi[p] = i;
    worst = bd[KNN-1];
  } else {
    int p = cnt++;
    while (p > 0 && bd[p-1] > d) { bd[p] = bd[p-1]; bi[p] = bi[p-1]; --p; }
    bd[p] = d; bi[p] = i;
    if (cnt == KNN) worst = bd[KNN-1];
  }
}

// ---------------- zero helper ----------------
__global__ void zero_f(float* p, int n) {
  int e = blockIdx.x * 256 + threadIdx.x;
  if (e < n) p[e] = 0.f;
}

// ---------------- 3-D KNN (scalar; K-dim is only 3) ----------------
__global__ void knn_d3(const float* __restrict__ xq, int qs,
                       const float* __restrict__ xs, int ss,
                       int nq, int ns,
                       const int* __restrict__ bq, const int* __restrict__ bs,
                       int excl, int* __restrict__ oidx, int* __restrict__ oval) {
  __shared__ float sp[128 * 3];
  __shared__ int   sb[128];
  int tid = threadIdx.x;
  int q = blockIdx.x * 128 + tid;
  bool act = q < nq;
  float q0 = 0.f, q1 = 0.f, q2 = 0.f; int myb = -2;
  if (act) { q0 = xq[q*qs]; q1 = xq[q*qs+1]; q2 = xq[q*qs+2]; myb = bq[q]; }
  float bd[KNN]; int bi[KNN]; int cnt = 0; float worst = __builtin_inff();
  for (int t = 0; t < ns; t += 128) {
    int c = t + tid;
    if (c < ns) {
      sp[tid*3+0] = xs[c*ss+0]; sp[tid*3+1] = xs[c*ss+1]; sp[tid*3+2] = xs[c*ss+2];
      sb[tid] = bs[c];
    }
    __syncthreads();
    int lim = min(128, ns - t);
    if (act) {
      for (int j = 0; j < lim; ++j) {
        int ci = t + j;
        if (sb[j] != myb) continue;
        if (excl && ci == q) continue;
        float dx = q0 - sp[j*3], dy = q1 - sp[j*3+1], dz = q2 - sp[j*3+2];
        float d2 = fmaxf(dx*dx + dy*dy + dz*dz, 0.f);
        topk_insert(d2, ci, bd, bi, cnt, worst);
      }
    }
    __syncthreads();
  }
  if (act) {
    for (int k = 0; k < KNN; ++k) {
      oidx[q*KNN+k] = (k < cnt) ? bi[k] : 0;
      oval[q*KNN+k] = (k < cnt && bd[k] <= R2C) ? 1 : 0;
    }
  }
}

// ---------------- WMMA fragment load (row-major, LDS or global) ----------------
__device__ __forceinline__ v16h ldfrag(const _Float16* base, int kb) {
  v8h lo = *(const v8h*)(base + kb);
  v8h hi = *(const v8h*)(base + kb + 16);
  return __builtin_shufflevector(lo, hi, 0,1,2,3,4,5,6,7,8,9,10,11,12,13,14,15);
}

// ---------------- 64-D KNN with WMMA distance GEMM ----------------
// Block: 128 threads (4 waves), 128 queries, candidate tiles of 64.
// Query (A) fragments are loop-invariant: loaded ONCE from global into registers
// (4 x v16h per wave-lane). Per tile only B fragments come from LDS, and each B
// fragment feeds two independent accumulator chains (both M-subtiles), letting
// the scheduler cover WMMA->WMMA RAW hazards. Gram tile stored transposed so
// each lane writes contiguous b128 chunks; all 128 lanes then select top-32.
#define QPAD 72   // 64 + 8 halves row padding (bank rotation, keeps 16B align)
__global__ void knn64_wmma(const _Float16* __restrict__ X, const float* __restrict__ XN,
                           const int* __restrict__ batch, int npts,
                           int* __restrict__ oidx, int* __restrict__ oval) {
  __shared__ __align__(16) _Float16 Ss[64 * QPAD];
  __shared__ __align__(16) float GsT[64 * 128];   // [cand][query]
  __shared__ float SNs[64]; __shared__ int SBs[64];

  int tid = threadIdx.x, wave = tid >> 5, lane = tid & 31;
  int q0 = blockIdx.x * 128;

  int kb   = (lane >= 16) ? 8 : 0;
  int hi8  = (lane >= 16) ? 8 : 0;
  int ncol = lane & 15;
  int mlan = lane & 15;

  // hoisted query fragments: M-subtiles (wave) and (wave+4), two K-halves each
  const _Float16* qr0 = &X[(size_t)(q0 + wave * 16 + mlan) * 64];
  const _Float16* qr1 = &X[(size_t)(q0 + (wave + 4) * 16 + mlan) * 64];
  v16h a00 = ldfrag(qr0, kb);
  v16h a01 = ldfrag(qr0 + 32, kb);
  v16h a10 = ldfrag(qr1, kb);
  v16h a11 = ldfrag(qr1 + 32, kb);

  float bd[KNN]; int bi[KNN]; int cnt = 0; float worst = __builtin_inff();
  int   myq  = q0 + tid;
  float myqn = XN[myq];
  int   myb  = batch[myq];

  for (int t = 0; t < npts; t += 64) {
    for (int l = tid; l < 512; l += 128) {
      int r = l >> 3, c8 = (l & 7) * 8;
      *(v8h*)&Ss[r * QPAD + c8] = *(const v8h*)&X[(size_t)(t + r) * 64 + c8];
    }
    if (tid < 64) { SNs[tid] = XN[t + tid]; SBs[tid] = batch[t + tid]; }
    if (t + 64 < npts) __builtin_prefetch(&X[(size_t)(t + 64) * 64], 0, 0);
    __syncthreads();

    #pragma unroll
    for (int ns = 0; ns < 4; ++ns) {
      const _Float16* brow = &Ss[(ns * 16 + mlan) * QPAD];
      v16h b0 = ldfrag(brow, kb);
      v16h b1 = ldfrag(brow + 32, kb);
      v8f c0 = {}, c1 = {};
      c0 = __builtin_amdgcn_wmma_f32_16x16x32_f16(false, a00, false, b0, (short)0, c0, false, false);
      c1 = __builtin_amdgcn_wmma_f32_16x16x32_f16(false, a10, false, b0, (short)0, c1, false, false);
      c0 = __builtin_amdgcn_wmma_f32_16x16x32_f16(false, a01, false, b1, (short)0, c0, false, false);
      c1 = __builtin_amdgcn_wmma_f32_16x16x32_f16(false, a11, false, b1, (short)0, c1, false, false);
      // transposed stores: this lane's 8 outputs are contiguous query-rows
      float* gp = &GsT[(ns * 16 + ncol) * 128 + wave * 16 + hi8];
      v4f l0 = { c0[0], c0[1], c0[2], c0[3] };
      v4f h0 = { c0[4], c0[5], c0[6], c0[7] };
      *(v4f*)gp       = l0;
      *(v4f*)(gp + 4) = h0;
      v4f l1 = { c1[0], c1[1], c1[2], c1[3] };
      v4f h1 = { c1[4], c1[5], c1[6], c1[7] };
      *(v4f*)(gp + 64)     = l1;   // M-subtile (wave+4) => +64 query rows
      *(v4f*)(gp + 64 + 4) = h1;
    }
    __syncthreads();

    // all 128 lanes select for their query
    for (int j = 0; j < 64; ++j) {
      int ci = t + j;
      if (SBs[j] != myb) continue;
      if (ci == myq) continue;              // exclude_self (always on for edge_conv)
      float d2 = fmaxf(myqn + SNs[j] - 2.f * GsT[j * 128 + tid], 0.f);
      topk_insert(d2, ci, bd, bi, cnt, worst);
    }
    __syncthreads();
  }
  for (int k = 0; k < KNN; ++k) {
    oidx[myq * KNN + k] = (k < cnt) ? bi[k] : 0;
    oval[myq * KNN + k] = (k < cnt && bd[k] <= R2C) ? 1 : 0;
  }
}

// ---------------- bidgcn: masked BN statistics (interior path, 6-d feats) -------
__global__ void msg_stats_i(const float* __restrict__ interior,
                            const int* __restrict__ idx, const int* __restrict__ val,
                            const float* __restrict__ W1, const float* __restrict__ b1,
                            float* __restrict__ stats /*129*/) {
  __shared__ float ssum[64], ssq[64]; __shared__ float scnt;
  int tid = threadIdx.x;
  if (tid < 64) { ssum[tid] = 0.f; ssq[tid] = 0.f; }
  if (tid == 0) scnt = 0.f;
  __syncthreads();
  int e = blockIdx.x * 256 + tid;
  if (e < NI * KNN && val[e]) {
    int n = e >> 5; int j = idx[e];
    float f0 = interior[n*3], f1 = interior[n*3+1], f2 = interior[n*3+2];
    float f3 = interior[j*3] - f0, f4 = interior[j*3+1] - f1, f5 = interior[j*3+2] - f2;
    atomicAdd(&scnt, 1.f);
    for (int c = 0; c < 64; ++c) {
      float y = b1[c] + f0*W1[c] + f1*W1[64+c] + f2*W1[128+c]
                      + f3*W1[192+c] + f4*W1[256+c] + f5*W1[320+c];
      atomicAdd(&ssum[c], y); atomicAdd(&ssq[c], y*y);
    }
  }
  __syncthreads();
  if (tid < 64) { atomicAdd(&stats[tid], ssum[tid]); atomicAdd(&stats[64+tid], ssq[tid]); }
  if (tid == 0) atomicAdd(&stats[128], scnt);
}

// ---------------- bidgcn stats (boundary path, 8-d feats) ----------------
__global__ void msg_stats_b(const float* __restrict__ interior, const float* __restrict__ boundary,
                            const int* __restrict__ idx, const int* __restrict__ val,
                            const float* __restrict__ W1, const float* __restrict__ b1,
                            float* __restrict__ stats /*129*/) {
  __shared__ float ssum[64], ssq[64]; __shared__ float scnt;
  int tid = threadIdx.x;
  if (tid < 64) { ssum[tid] = 0.f; ssq[tid] = 0.f; }
  if (tid == 0) scnt = 0.f;
  __syncthreads();
  int e = blockIdx.x * 256 + tid;
  if (e < NI * KNN && val[e]) {
    int n = e >> 5; int j = idx[e];
    float f0 = interior[n*3], f1 = interior[n*3+1], f2 = interior[n*3+2];
    float f3 = boundary[j*5]   - f0;
    float f4 = boundary[j*5+1] - f1;
    float f5 = boundary[j*5+2] - f2;
    float f6 = boundary[j*5+3], f7 = boundary[j*5+4];
    atomicAdd(&scnt, 1.f);
    for (int c = 0; c < 64; ++c) {
      float y = b1[c] + f0*W1[c] + f1*W1[64+c] + f2*W1[128+c] + f3*W1[192+c]
                      + f4*W1[256+c] + f5*W1[320+c] + f6*W1[384+c] + f7*W1[448+c];
      atomicAdd(&ssum[c], y); atomicAdd(&ssq[c], y*y);
    }
  }
  __syncthreads();
  if (tid < 64) { atomicAdd(&stats[tid], ssum[tid]); atomicAdd(&stats[64+tid], ssq[tid]); }
  if (tid == 0) atomicAdd(&stats[128], scnt);
}

// ---------------- bidgcn apply (interior): BN->relu->sum_k, then @W2 collapse ----
__global__ void msg_apply_i(const float* __restrict__ interior,
                            const int* __restrict__ idx, const int* __restrict__ val,
                            const float* __restrict__ W1, const float* __restrict__ b1,
                            const float* __restrict__ g, const float* __restrict__ be,
                            const float* __restrict__ W2, const float* __restrict__ b2,
                            const float* __restrict__ stats,
                            float* __restrict__ si, float* __restrict__ ci) {
  __shared__ float xj[KNN * 3]; __shared__ int vf[KNN]; __shared__ float sA[64];
  __shared__ float xi[3];
  int n = blockIdx.x, c = threadIdx.x;  // 64 threads
  if (c < 3) xi[c] = interior[n*3+c];
  if (c < KNN) {
    int e = n*KNN + c; int v = val[e]; vf[c] = v;
    if (v) { int j = idx[e]; xj[c*3] = interior[j*3]; xj[c*3+1] = interior[j*3+1]; xj[c*3+2] = interior[j*3+2]; }
  }
  __syncthreads();
  float cT = fmaxf(stats[128], 1.f);
  float mean = stats[c] / cT;
  float var  = fmaxf(stats[64+c] / cT - mean*mean, 0.f);
  float scale = rsqrtf(var + EPSV) * g[c], shift = be[c];
  float w0 = W1[c], w1 = W1[64+c], w2 = W1[128+c], w3 = W1[192+c], w4 = W1[256+c], w5 = W1[320+c];
  float acc = 0.f; int pc = 0;
  for (int k = 0; k < KNN; ++k) {
    if (!vf[k]) continue;
    ++pc;
    float f3 = xj[k*3] - xi[0], f4 = xj[k*3+1] - xi[1], f5 = xj[k*3+2] - xi[2];
    float y = b1[c] + xi[0]*w0 + xi[1]*w1 + xi[2]*w2 + f3*w3 + f4*w4 + f5*w5;
    acc += fmaxf((y - mean) * scale + shift, 0.f);
  }
  sA[c] = acc;
  __syncthreads();
  float cntf = (float)pc;
  float out = cntf * b2[c];
  for (int j = 0; j < 64; ++j) out += sA[j] * W2[j*64+c];
  si[n*64+c] = out;
  if (c == 0) ci[n] = cntf;
}

// ---------------- bidgcn apply (boundary) ----------------
__global__ void msg_apply_b(const float* __restrict__ interior, const float* __restrict__ boundary,
                            const int* __restrict__ idx, const int* __restrict__ val,
                            const float* __restrict__ W1, const float* __restrict__ b1,
                            const float* __restrict__ g, const float* __restrict__ be,
                            const float* __restrict__ W2, const float* __restrict__ b2,
                            const float* __restrict__ stats,
                            float* __restrict__ sb, float* __restrict__ cb) {
  __shared__ float xb[KNN * 5]; __shared__ int vf[KNN]; __shared__ float sA[64];
  __shared__ float xi[3];
  int n = blockIdx.x, c = threadIdx.x;  // 64 threads
  if (c < 3) xi[c] = interior[n*3+c];
  if (c < KNN) {
    int e = n*KNN + c; int v = val[e]; vf[c] = v;
    if (v) { int j = idx[e];
      xb[c*5]=boundary[j*5]; xb[c*5+1]=boundary[j*5+1]; xb[c*5+2]=boundary[j*5+2];
      xb[c*5+3]=boundary[j*5+3]; xb[c*5+4]=boundary[j*5+4]; }
  }
  __syncthreads();
  float cT = fmaxf(stats[128], 1.f);
  float mean = stats[c] / cT;
  float var  = fmaxf(stats[64+c] / cT - mean*mean, 0.f);
  float scale = rsqrtf(var + EPSV) * g[c], shift = be[c];
  float w0=W1[c],w1=W1[64+c],w2=W1[128+c],w3=W1[192+c],w4=W1[256+c],w5=W1[320+c],w6=W1[384+c],w7=W1[448+c];
  float acc = 0.f; int pc = 0;
  for (int k = 0; k < KNN; ++k) {
    if (!vf[k]) continue;
    ++pc;
    float f3 = xb[k*5]-xi[0], f4 = xb[k*5+1]-xi[1], f5 = xb[k*5+2]-xi[2];
    float y = b1[c] + xi[0]*w0 + xi[1]*w1 + xi[2]*w2 + f3*w3 + f4*w4 + f5*w5
                    + xb[k*5+3]*w6 + xb[k*5+4]*w7;
    acc += fmaxf((y - mean) * scale + shift, 0.f);
  }
  sA[c] = acc;
  __syncthreads();
  float cntf = (float)pc;
  float out = cntf * b2[c];
  for (int j = 0; j < 64; ++j) out += sA[j] * W2[j*64+c];
  sb[n*64+c] = out;
  if (c == 0) cb[n] = cntf;
}

// ---------------- h0 = (si+sb)/max(ci+cb,1) -> xcat block 0 ----------------
__global__ void combine_h0(const float* __restrict__ si, const float* __restrict__ ci,
                           const float* __restrict__ sb, const float* __restrict__ cb,
                           float* __restrict__ xcat) {
  int e = blockIdx.x * 256 + threadIdx.x;
  if (e >= NI * 64) return;
  int n = e >> 6, c = e & 63;
  float denom = fmaxf(ci[n] + cb[n], 1.f);
  xcat[n*320 + c] = (si[e] + sb[e]) / denom;
}

// ---------------- in-place relu on one 64-col block + f16 copy + row norm -------
__global__ void relu_pack_norm(float* __restrict__ xcat, int coff,
                               _Float16* __restrict__ xh, float* __restrict__ xn) {
  __shared__ float red[64];
  int n = blockIdx.x, c = threadIdx.x;  // 64 threads
  float v = fmaxf(xcat[n*320 + coff + c], 0.f);
  xcat[n*320 + coff + c] = v;
  xh[n*64 + c] = (_Float16)v;
  red[c] = v * v;
  __syncthreads();
  for (int s = 32; s > 0; s >>= 1) { if (c < s) red[c] += red[c+s]; __syncthreads(); }
  if (c == 0) xn[n] = red[0];
}

// ---------------- edge_conv collapsed: [cnt*xi, sum(xj)-cnt*xi] @ W + cnt*b -----
__global__ void edge_reduce(float* __restrict__ xcat, int cin, int cout,
                            const int* __restrict__ idx, const int* __restrict__ val,
                            const float* __restrict__ W, const float* __restrict__ b) {
  __shared__ float sA[128]; __shared__ int sidx[KNN]; __shared__ int sval[KNN];
  int n = blockIdx.x, c = threadIdx.x;  // 64 threads
  if (c < KNN) { sidx[c] = idx[n*KNN+c]; sval[c] = val[n*KNN+c]; }
  __syncthreads();
  float xi = xcat[n*320 + cin + c];
  float gsum = 0.f; int cnt = 0;
  for (int k = 0; k < KNN; ++k) {
    if (!sval[k]) continue;
    ++cnt;
    gsum += xcat[sidx[k]*320 + cin + c];
  }
  float cntf = (float)cnt;
  sA[c] = cntf * xi;
  sA[64 + c] = gsum - cntf * xi;
  __syncthreads();
  float out = cntf * b[c];
  for (int j = 0; j < 128; ++j) out += sA[j] * W[j*64+c];
  xcat[n*320 + cout + c] = out / fmaxf(cntf, 1.f);
}

// ---------------- f32 -> f16 elementwise ----------------
__global__ void pack_f16(const float* __restrict__ x, _Float16* __restrict__ xh, int n) {
  int e = blockIdx.x * 256 + threadIdx.x;
  if (e < n) xh[e] = (_Float16)x[e];
}

// ---------------- WMMA tiled GEMM: C = A(MxK,f16) * B(KxN,f16) + bias ----------
// Block 256 thr (8 waves), tile 128x128, K-step 32. B transposed into LDS.
#define GPAD 40   // 32 + 8 halves padding
__global__ void wmma_gemm(const _Float16* __restrict__ A, const _Float16* __restrict__ B,
                          const float* __restrict__ bias, float* __restrict__ C,
                          int M, int Nn, int Kd) {
  __shared__ __align__(16) _Float16 As[128 * GPAD];
  __shared__ __align__(16) _Float16 Bs[128 * GPAD];  // [n][k]
  int tid = threadIdx.x, wave = tid >> 5, lane = tid & 31;
  int bm = blockIdx.x * 128, bn = blockIdx.y * 128;
  v8f acc[8] = {};
  int mrow = wave * 16 + (lane & 15);
  int kb = (lane >= 16) ? 8 : 0;
  for (int k0 = 0; k0 < Kd; k0 += 32) {
    for (int l = tid; l < 512; l += 256) {
      int r = l >> 2, c8 = (l & 3) * 8;
      *(v8h*)&As[r * GPAD + c8] = *(const v8h*)&A[(size_t)(bm + r) * Kd + k0 + c8];
    }
    for (int l = tid; l < 512; l += 256) {
      int kk = l >> 4, n0 = (l & 15) * 8;
      v8h v = *(const v8h*)&B[(size_t)(k0 + kk) * Nn + bn + n0];
      #pragma unroll
      for (int i = 0; i < 8; ++i) Bs[(n0 + i) * GPAD + kk] = v[i];
    }
    if (k0 + 32 < Kd) __builtin_prefetch(&A[(size_t)bm * Kd + k0 + 32], 0, 0);
    __syncthreads();
    v16h a = ldfrag(&As[mrow * GPAD], kb);
    #pragma unroll
    for (int ns = 0; ns < 8; ++ns) {
      int nrow = ns * 16 + (lane & 15);
      v16h bfr = ldfrag(&Bs[nrow * GPAD], kb);
      acc[ns] = __builtin_amdgcn_wmma_f32_16x16x32_f16(false, a, false, bfr, (short)0, acc[ns], false, false);
    }
    __syncthreads();
  }
  int hi8 = (lane >= 16) ? 8 : 0;
  int ncol = lane & 15;
  #pragma unroll
  for (int ns = 0; ns < 8; ++ns) {
    #pragma unroll
    for (int r = 0; r < 8; ++r) {
      int gr = bm + wave * 16 + r + hi8;
      int gc = bn + ns * 16 + ncol;
      C[(size_t)gr * Nn + gc] = acc[ns][r] + bias[gc];
    }
  }
}

// ---------------- per-column sum/sumsq over M rows (N=256) ----------------
__global__ void col_stats(const float* __restrict__ Y, float* __restrict__ stats, int M) {
  __shared__ float s1[256], s2[256];
  int c = blockIdx.x, tid = threadIdx.x;  // 256 threads
  float a = 0.f, b = 0.f;
  for (int r = tid; r < M; r += 256) { float v = Y[(size_t)r * 256 + c]; a += v; b += v*v; }
  s1[tid] = a; s2[tid] = b;
  __syncthreads();
  for (int s = 128; s > 0; s >>= 1) {
    if (tid < s) { s1[tid] += s1[tid+s]; s2[tid] += s2[tid+s]; }
    __syncthreads();
  }
  if (tid == 0) { stats[c] = s1[0]; stats[256 + c] = s2[0]; }
}

// ---------------- bn1d + relu -> f16 ----------------
__global__ void bn_relu_pack(const float* __restrict__ Y, const float* __restrict__ stats,
                             const float* __restrict__ g, const float* __restrict__ be,
                             _Float16* __restrict__ out, int M) {
  int e = blockIdx.x * 256 + threadIdx.x;
  if (e >= M * 256) return;
  int c = e & 255;
  float inv = 1.f / (float)M;
  float mean = stats[c] * inv;
  float var = fmaxf(stats[256 + c] * inv - mean * mean, 0.f);
  float v = (Y[e] - mean) * rsqrtf(var + EPSV) * g[c] + be[c];
  out[e] = (_Float16)fmaxf(v, 0.f);
}

// ---------------- final: bn1d + relu fused with 256x2 head + sigmoid ------------
__global__ void final_head(const float* __restrict__ Y, const float* __restrict__ stats,
                           const float* __restrict__ g, const float* __restrict__ be,
                           const float* __restrict__ W3, const float* __restrict__ b3,
                           float* __restrict__ out, int M) {
  int n = blockIdx.x * 256 + threadIdx.x;
  if (n >= M) return;
  float inv = 1.f / (float)M;
  float a0 = b3[0], a1 = b3[1];
  for (int c = 0; c < 256; ++c) {
    float mean = stats[c] * inv;
    float var = fmaxf(stats[256 + c] * inv - mean * mean, 0.f);
    float v = (Y[(size_t)n * 256 + c] - mean) * rsqrtf(var + EPSV) * g[c] + be[c];
    v = fmaxf(v, 0.f);
    a0 += v * W3[c*2]; a1 += v * W3[c*2+1];
  }
  out[n*2+0] = 1.f / (1.f + expf(-a0));
  out[n*2+1] = 1.f / (1.f + expf(-a1));
}

// =============================== host launcher ===============================
extern "C" void kernel_launch(void* const* d_in, const int* in_sizes, int n_in,
                              void* d_out, int out_size, void* d_ws, size_t ws_size,
                              hipStream_t stream) {
  (void)in_sizes; (void)n_in; (void)out_size; (void)ws_size;
  const float* interior  = (const float*)d_in[0];
  const float* boundary  = (const float*)d_in[1];
  const int*   batch_int = (const int*)d_in[2];
  const int*   batch_bdr = (const int*)d_in[3];
  const float* mi_W1 = (const float*)d_in[4];  const float* mi_b1 = (const float*)d_in[5];
  const float* mi_g  = (const float*)d_in[6];  const float* mi_be = (const float*)d_in[7];
  const float* mi_W2 = (const float*)d_in[8];  const float* mi_b2 = (const float*)d_in[9];
  const float* mb_W1 = (const float*)d_in[10]; const float* mb_b1 = (const float*)d_in[11];
  const float* mb_g  = (const float*)d_in[12]; const float* mb_be = (const float*)d_in[13];
  const float* mb_W2 = (const float*)d_in[14]; const float* mb_b2 = (const float*)d_in[15];
  const float* o_W1  = (const float*)d_in[16]; const float* o_b1  = (const float*)d_in[17];
  const float* o_g1  = (const float*)d_in[18]; const float* o_be1 = (const float*)d_in[19];
  const float* o_W2  = (const float*)d_in[20]; const float* o_b2  = (const float*)d_in[21];
  const float* o_g2  = (const float*)d_in[22]; const float* o_be2 = (const float*)d_in[23];
  const float* o_W3  = (const float*)d_in[24]; const float* o_b3  = (const float*)d_in[25];
  const float* hW[4] = {(const float*)d_in[26], (const float*)d_in[28], (const float*)d_in[30], (const float*)d_in[32]};
  const float* hb[4] = {(const float*)d_in[27], (const float*)d_in[29], (const float*)d_in[31], (const float*)d_in[33]};

  char* ws = (char*)d_ws;
  size_t off = 0;
  auto alloc = [&](size_t bytes) -> void* {
    off = (off + 255) & ~(size_t)255;
    void* p = ws + off;
    off += bytes;
    return p;
  };

  int*      idxI   = (int*)alloc((size_t)NI * KNN * 4);
  int*      valI   = (int*)alloc((size_t)NI * KNN * 4);
  int*      idxB   = (int*)alloc((size_t)NI * KNN * 4);
  int*      valB   = (int*)alloc((size_t)NI * KNN * 4);
  float*    statsAll = (float*)alloc(1536 * 4);
  float*    statsMi = statsAll;                          // 129 floats
  float*    statsMb = statsAll + 192;                    // 129 floats
  float*    ostats1 = statsAll + 512;                    // 512 floats
  float*    ostats2 = statsAll + 1024;                   // 512 floats
  float*    si = (float*)alloc((size_t)NI * 64 * 4);
  float*    ci = (float*)alloc((size_t)NI * 4);
  float*    sbuf = (float*)alloc((size_t)NI * 64 * 4);
  float*    cbuf = (float*)alloc((size_t)NI * 4);
  float*    xcat = (float*)alloc((size_t)NI * 320 * 4);
  _Float16* xh16 = (_Float16*)alloc((size_t)NI * 64 * 2);
  float*    xnorm = (float*)alloc((size_t)NI * 4);
  _Float16* Xh16 = (_Float16*)alloc((size_t)NI * 320 * 2);
  _Float16* w1h = (_Float16*)alloc((size_t)320 * 256 * 2);
  _Float16* w2h = (_Float16*)alloc((size_t)256 * 256 * 2);
  float*    Y   = (float*)alloc((size_t)NI * 256 * 4);
  _Float16* A2h = (_Float16*)alloc((size_t)NI * 256 * 2);

  // 0. zero accumulators
  zero_f<<<(1536 + 255) / 256, 256, 0, stream>>>(statsAll, 1536);

  // 1. 3-D KNN graphs
  knn_d3<<<NI / 128, 128, 0, stream>>>(interior, 3, interior, 3, NI, NI,
                                       batch_int, batch_int, 1, idxI, valI);
  knn_d3<<<NI / 128, 128, 0, stream>>>(interior, 3, boundary, 5, NI, NBD,
                                       batch_int, batch_bdr, 0, idxB, valB);

  // 2. bidgcn: masked BN stats, apply, combine -> h0 (xcat cols 0..63)
  msg_stats_i<<<NI * KNN / 256, 256, 0, stream>>>(interior, idxI, valI, mi_W1, mi_b1, statsMi);
  msg_stats_b<<<NI * KNN / 256, 256, 0, stream>>>(interior, boundary, idxB, valB, mb_W1, mb_b1, statsMb);
  msg_apply_i<<<NI, 64, 0, stream>>>(interior, idxI, valI, mi_W1, mi_b1, mi_g, mi_be,
                                     mi_W2, mi_b2, statsMi, si, ci);
  msg_apply_b<<<NI, 64, 0, stream>>>(interior, boundary, idxB, valB, mb_W1, mb_b1, mb_g, mb_be,
                                     mb_W2, mb_b2, statsMb, sbuf, cbuf);
  combine_h0<<<NI * 64 / 256, 256, 0, stream>>>(si, ci, sbuf, cbuf, xcat);

  // 3. four edge_conv layers: relu -> 64-D WMMA KNN -> collapsed gather-matvec
  for (int i = 0; i < 4; ++i) {
    relu_pack_norm<<<NI, 64, 0, stream>>>(xcat, 64 * i, xh16, xnorm);
    knn64_wmma<<<NI / 128, 128, 0, stream>>>(xh16, xnorm, batch_int, NI, idxI, valI);
    edge_reduce<<<NI, 64, 0, stream>>>(xcat, 64 * i, 64 * (i + 1), idxI, valI, hW[i], hb[i]);
  }

  // 4. final MLP with WMMA GEMMs
  pack_f16<<<NI * 320 / 256, 256, 0, stream>>>(xcat, Xh16, NI * 320);
  pack_f16<<<320 * 256 / 256, 256, 0, stream>>>(o_W1, w1h, 320 * 256);
  pack_f16<<<256 * 256 / 256, 256, 0, stream>>>(o_W2, w2h, 256 * 256);

  wmma_gemm<<<dim3(NI / 128, 2), 256, 0, stream>>>(Xh16, w1h, o_b1, Y, NI, 256, 320);
  col_stats<<<256, 256, 0, stream>>>(Y, ostats1, NI);
  bn_relu_pack<<<NI * 256 / 256, 256, 0, stream>>>(Y, ostats1, o_g1, o_be1, A2h, NI);

  wmma_gemm<<<dim3(NI / 128, 2), 256, 0, stream>>>(A2h, w2h, o_b2, Y, NI, 256, 256);
  col_stats<<<256, 256, 0, stream>>>(Y, ostats2, NI);
  final_head<<<(NI + 255) / 256, 256, 0, stream>>>(Y, ostats2, o_g2, o_be2, o_W3, o_b3,
                                                   (float*)d_out, NI);
}
